// MultilevelEncoder_13907104104765
// MI455X (gfx1250) — matrix operand order repaired and checked by reading, not compiled, not run on hardware
//
#include <hip/hip_runtime.h>
#include <hip/hip_bf16.h>

// ---------------------------------------------------------------------------
// MultilevelEncoder for MI455X (gfx1250, wave32, WMMA bf16 path)
// B=64, L=256, D=2048, E=1024, WIN=3
// ---------------------------------------------------------------------------

#define BB  64
#define LL  256
#define DD  2048
#define EE  1024
#define LW  254           // L - WIN + 1
#define NEGF (-1e18f)

typedef __bf16 v16bf __attribute__((ext_vector_type(16)));
typedef __bf16 v8bf  __attribute__((ext_vector_type(8)));
typedef __bf16 bf4   __attribute__((ext_vector_type(4)));
typedef float  v8f   __attribute__((ext_vector_type(8)));

__device__ __forceinline__ float gelu_erf(float x) {
  return 0.5f * x * (1.0f + erff(x * 0.70710678118654752f));
}

// --- CDNA5 async global->LDS copy (ASYNCcnt-tracked DMA, bypasses VGPRs) ---
__device__ __forceinline__ void async_ld_b128(unsigned int lds_byte, const void* g) {
  asm volatile("global_load_async_to_lds_b128 %0, %1, off"
               :
               : "v"(lds_byte), "v"((unsigned long long)(uintptr_t)g)
               : "memory");
}
__device__ __forceinline__ void wait_async0() {
  asm volatile("s_wait_asynccnt 0x0" ::: "memory");
}

// ---------------------------------------------------------------------------
// 1) mot_scores = clip_ft @ w_mot_wei + b, masked with pad -> NEG
//    one wave per (b,l) row, D=2048 reduction
// ---------------------------------------------------------------------------
__global__ __launch_bounds__(256)
void scores_kernel(const float* __restrict__ clip, const float* __restrict__ w,
                   const float* __restrict__ bsc, const int* __restrict__ lens,
                   float* __restrict__ out_scores) {
  const int lane = threadIdx.x & 31;
  const int warp = threadIdx.x >> 5;
  const int r = blockIdx.x * 8 + warp;            // row in [0, B*L)
  const float* row = clip + (size_t)r * DD;
  float s = 0.f;
  #pragma unroll 4
  for (int i = lane; i < DD; i += 32) s += row[i] * w[i];
  #pragma unroll
  for (int m = 16; m; m >>= 1) s += __shfl_xor(s, m, 32);
  if (lane == 0) {
    const int b = r >> 8, l = r & 255;
    float v = s + bsc[0];
    if (l >= lens[b]) v = NEGF;
    out_scores[r] = v;
  }
}

// ---------------------------------------------------------------------------
// 2) sliding-window (size 3) softmax weights
// ---------------------------------------------------------------------------
__global__ __launch_bounds__(256)
void win_softmax_kernel(const float* __restrict__ scores, float* __restrict__ awin) {
  const int idx = blockIdx.x * 256 + threadIdx.x;  // over B*LW
  if (idx >= BB * LW) return;
  const int b = idx / LW, l = idx % LW;
  const float s0 = scores[b * LL + l + 0];
  const float s1 = scores[b * LL + l + 1];
  const float s2 = scores[b * LL + l + 2];
  const float m = fmaxf(s0, fmaxf(s1, s2));
  const float e0 = expf(s0 - m), e1 = expf(s1 - m), e2 = expf(s2 - m);
  const float inv = 1.f / (e0 + e1 + e2);
  awin[idx * 3 + 0] = e0 * inv;
  awin[idx * 3 + 1] = e1 * inv;
  awin[idx * 3 + 2] = e2 * inv;
}

// ---------------------------------------------------------------------------
// 3) clip_ft f32 -> bf16
// ---------------------------------------------------------------------------
__global__ __launch_bounds__(256)
void conv_clip_kernel(const float* __restrict__ clip, __bf16* __restrict__ clipb) {
  const size_t i4 = (size_t)blockIdx.x * 256 + threadIdx.x;  // over B*L*D/4
  const float4 v = ((const float4*)clip)[i4];
  bf4 o;
  o.x = (__bf16)v.x; o.y = (__bf16)v.y; o.z = (__bf16)v.z; o.w = (__bf16)v.w;
  ((bf4*)clipb)[i4] = o;
}

// ---------------------------------------------------------------------------
// 4) vid_clip[b,l,d] = a0*c[l] + a1*c[l+1] + a2*c[l+2]  (bf16 output)
//    clip_ft (134MB) fits in the 192MB L2, so the 3x reads mostly hit L2.
// ---------------------------------------------------------------------------
__global__ __launch_bounds__(256)
void conv_vid_kernel(const float* __restrict__ clip, const float* __restrict__ awin,
                     __bf16* __restrict__ vidb) {
  const size_t i = (size_t)blockIdx.x * 256 + threadIdx.x;  // over B*LW*(D/4)
  const int d4 = (int)(i % (DD / 4));
  const size_t bl = i / (DD / 4);
  const int b = (int)(bl / LW), l = (int)(bl % LW);
  const float a0 = awin[bl * 3 + 0];
  const float a1 = awin[bl * 3 + 1];
  const float a2 = awin[bl * 3 + 2];
  const float* base = clip + ((size_t)b * LL + l) * DD + d4 * 4;
  const float4 c0 = *(const float4*)(base);
  const float4 c1 = *(const float4*)(base + DD);
  const float4 c2 = *(const float4*)(base + 2 * DD);
  bf4 o;
  o.x = (__bf16)(a0 * c0.x + a1 * c1.x + a2 * c2.x);
  o.y = (__bf16)(a0 * c0.y + a1 * c1.y + a2 * c2.y);
  o.z = (__bf16)(a0 * c0.z + a1 * c1.z + a2 * c2.z);
  o.w = (__bf16)(a0 * c0.w + a1 * c1.w + a2 * c2.w);
  ((bf4*)vidb)[bl * (DD / 4) + d4] = o;
}

// ---------------------------------------------------------------------------
// 5) weight transpose + convert: w[k][n] f32 (DxE) -> wT[n][k] bf16 (ExD)
//    so GEMM B-tiles load with contiguous K exactly like A-tiles.
// ---------------------------------------------------------------------------
__global__ __launch_bounds__(256)
void convw_kernel(const float* __restrict__ w, __bf16* __restrict__ wT) {
  __shared__ float tile[32][33];
  const int tx = threadIdx.x & 31;
  const int ty = threadIdx.x >> 5;      // 0..7
  const int n0 = blockIdx.x * 32;
  const int k0 = blockIdx.y * 32;
  #pragma unroll
  for (int j = 0; j < 4; ++j) {
    const int k = k0 + ty + j * 8;
    tile[ty + j * 8][tx] = w[(size_t)k * EE + n0 + tx];
  }
  __syncthreads();
  #pragma unroll
  for (int j = 0; j < 4; ++j) {
    const int n = n0 + ty + j * 8;
    wT[(size_t)n * DD + k0 + tx] = (__bf16)tile[tx][ty + j * 8];
  }
}

// ---------------------------------------------------------------------------
// 6) bf16 WMMA GEMM: out[M x E] = gelu(A[M x D] * BT[E x D]^T + bias)
//    128x128 block tile, 8 waves (2 x 4), wave tile 64x32 (4x2 wmma accs).
//    Software-pipelined with CDNA5 async global->LDS DMA: copies for tile
//    k+1 are issued (ASYNCcnt) before the WMMA block of tile k, then
//    s_wait_asynccnt 0 + one barrier per K-step. LDS ping-pong buffers.
// ---------------------------------------------------------------------------
#define LDSK 40   // 32 + 8 pad, keeps 16B alignment (80B row stride)

__global__ __launch_bounds__(256)
void gemm_bf16_kernel(const __bf16* __restrict__ A, const __bf16* __restrict__ BT,
                      const float* __restrict__ bias, float* __restrict__ out) {
  __shared__ __bf16 As[2][128 * LDSK];
  __shared__ __bf16 Bs[2][128 * LDSK];

  const int tid  = threadIdx.x;
  const int lane = tid & 31;
  const int wv   = tid >> 5;         // 0..7
  const int wm   = wv >> 2;          // 0..1  (M)
  const int wn   = wv & 3;           // 0..3  (N)
  const int mBase = blockIdx.y * 128;
  const int nBase = blockIdx.x * 128;

  // cooperative tile loaders: thread -> (row 0..127, k-half 0..1), 32B each
  const int ldr = tid >> 1;
  const int ldh = tid & 1;
  const __bf16* aG = A  + (size_t)(mBase + ldr) * DD + ldh * 16;
  const __bf16* bG = BT + (size_t)(nBase + ldr) * DD + ldh * 16;
  const int ldsOff = ldr * LDSK + ldh * 16;

  // per-thread LDS byte addresses of this thread's staging slot, both buffers
  const unsigned int aL0 = (unsigned int)(uintptr_t)&As[0][ldsOff];
  const unsigned int aL1 = (unsigned int)(uintptr_t)&As[1][ldsOff];
  const unsigned int bL0 = (unsigned int)(uintptr_t)&Bs[0][ldsOff];
  const unsigned int bL1 = (unsigned int)(uintptr_t)&Bs[1][ldsOff];

  v8f acc[4][2] = {};

  const int lr = lane & 15;          // row/col within 16
  const int hi = lane >> 4;          // K-group select

  // preload tile 0 into buffer 0 via async DMA
  async_ld_b128(aL0,      aG);
  async_ld_b128(aL0 + 16, aG + 8);
  async_ld_b128(bL0,      bG);
  async_ld_b128(bL0 + 16, bG + 8);
  __builtin_prefetch(aG + 32, 0, 1);   // global_prefetch_b8 (speculative)
  __builtin_prefetch(bG + 32, 0, 1);
  wait_async0();
  __syncthreads();

  int p = 0;
  for (int it = 0; it < DD / 32; ++it) {
    const int kn = (it + 1) * 32;
    const bool more = kn < DD;

    // kick off next tile's DMA into the other buffer before this tile's math
    if (more) {
      const unsigned int aDst = p ? aL0 : aL1;
      const unsigned int bDst = p ? bL0 : bL1;
      async_ld_b128(aDst,      aG + kn);
      async_ld_b128(aDst + 16, aG + kn + 8);
      async_ld_b128(bDst,      bG + kn);
      async_ld_b128(bDst + 16, bG + kn + 8);
      __builtin_prefetch(aG + kn + 64, 0, 1);  // two tiles ahead (speculative)
      __builtin_prefetch(bG + kn + 64, 0, 1);
    }

    // ---- compute on buffer p ----
    const __bf16* AsP = As[p];
    const __bf16* BsP = Bs[p];

    // A fragment (16x32 bf16): lane lr = row M, VGPR0-3 hold K (hi*8..),
    // VGPR4-7 hold K (16 + hi*8..) per ISA 7.12.2.
    v16bf afr[4];
    #pragma unroll
    for (int mi = 0; mi < 4; ++mi) {
      const __bf16* q = &AsP[(wm * 64 + mi * 16 + lr) * LDSK];
      const v8bf lo = *(const v8bf*)(q + hi * 8);
      const v8bf hh = *(const v8bf*)(q + 16 + hi * 8);
      afr[mi] = __builtin_shufflevector(lo, hh, 0, 1, 2, 3, 4, 5, 6, 7,
                                        8, 9, 10, 11, 12, 13, 14, 15);
    }
    // B fragment (32x16 bf16): lane lr = col N, K (hi*8..) then K (16 + hi*8..).
    v16bf bfr[2];
    #pragma unroll
    for (int ni = 0; ni < 2; ++ni) {
      const __bf16* q = &BsP[(wn * 32 + ni * 16 + lr) * LDSK];
      const v8bf lo = *(const v8bf*)(q + hi * 8);
      const v8bf hh = *(const v8bf*)(q + 16 + hi * 8);
      bfr[ni] = __builtin_shufflevector(lo, hh, 0, 1, 2, 3, 4, 5, 6, 7,
                                        8, 9, 10, 11, 12, 13, 14, 15);
    }
    #pragma unroll
    for (int mi = 0; mi < 4; ++mi)
      #pragma unroll
      for (int ni = 0; ni < 2; ++ni)
        acc[mi][ni] = __builtin_amdgcn_wmma_f32_16x16x32_bf16(
            false, afr[mi], false, bfr[ni], (short)0, acc[mi][ni], false, false);

    // ---- flip buffers: own DMA done, then block-wide visibility ----
    if (more) {
      wait_async0();
      __syncthreads();
      p ^= 1;
    }
  }

  // epilogue: bias + exact-erf GELU, f32 store
  #pragma unroll
  for (int mi = 0; mi < 4; ++mi) {
    #pragma unroll
    for (int ni = 0; ni < 2; ++ni) {
      const int col = nBase + wn * 32 + ni * 16 + lr;
      const float bv = bias[col];
      #pragma unroll
      for (int r = 0; r < 8; ++r) {
        const int row = mBase + wm * 64 + mi * 16 + r + 8 * hi;  // C/D layout
        out[(size_t)row * EE + col] = gelu_erf(acc[mi][ni][r] + bv);
      }
    }
  }
}

// ---------------------------------------------------------------------------
// 7) masked-softmax attention pooling; one block per (batch, branch)
//    branch 0: motion (rows=LW, len-2), branch 1: appearance (rows=L, len)
// ---------------------------------------------------------------------------
__global__ __launch_bounds__(256)
void pool_kernel(const float* __restrict__ motion, const float* __restrict__ appear,
                 const float* __restrict__ wpool, const float* __restrict__ bpool,
                 const int* __restrict__ lens, float* __restrict__ out) {
  const int b = blockIdx.x;
  const int which = blockIdx.y;
  const float* src;
  int rows, len_eff;
  if (which == 0) { src = motion + (size_t)b * LW * EE; rows = LW; len_eff = lens[b] - 2; }
  else            { src = appear + (size_t)b * LL * EE; rows = LL; len_eff = lens[b]; }

  __shared__ float sA[256];
  __shared__ float red[8];
  const int tid = threadIdx.x, lane = tid & 31, warp = tid >> 5;

  // phase 1: s[l] = gelu(dot(src[l,:], w_pool) + b_pool), wave per row
  for (int l = warp; l < rows; l += 8) {
    const float* rp = src + (size_t)l * EE;
    float s = 0.f;
    #pragma unroll 4
    for (int i = lane; i < EE; i += 32) s += rp[i] * wpool[i];
    #pragma unroll
    for (int m = 16; m; m >>= 1) s += __shfl_xor(s, m, 32);
    if (lane == 0) sA[l] = gelu_erf(s + bpool[0]);
  }
  __syncthreads();

  // phase 2: masked softmax over rows
  const float x = (tid < rows && tid < len_eff) ? sA[tid] : NEGF;
  float m = x;
  #pragma unroll
  for (int s2 = 16; s2; s2 >>= 1) m = fmaxf(m, __shfl_xor(m, s2, 32));
  if (lane == 0) red[warp] = m;
  __syncthreads();
  float mx = red[0];
  #pragma unroll
  for (int i = 1; i < 8; ++i) mx = fmaxf(mx, red[i]);
  const float e = (tid < rows && tid < len_eff) ? expf(x - mx) : 0.f;
  float ss = e;
  #pragma unroll
  for (int s2 = 16; s2; s2 >>= 1) ss += __shfl_xor(ss, s2, 32);
  __syncthreads();
  if (lane == 0) red[warp] = ss;
  __syncthreads();
  float tot = 0.f;
  #pragma unroll
  for (int i = 0; i < 8; ++i) tot += red[i];
  const float a = e / tot;
  __syncthreads();
  sA[tid] = a;                      // masked rows get 0
  __syncthreads();

  // phase 3: out[e] = sum_l a[l] * src[l,e]; 4 columns per thread
  float accv[4] = {0.f, 0.f, 0.f, 0.f};
  for (int l = 0; l < rows; ++l) {
    const float al = sA[l];
    const float* rp = src + (size_t)l * EE;
    #pragma unroll
    for (int j = 0; j < 4; ++j) accv[j] += al * rp[tid + j * 256];
  }
  float* op = out + (size_t)b * (2 * EE) + which * EE;
  #pragma unroll
  for (int j = 0; j < 4; ++j) op[tid + j * 256] = accv[j];
}

// ---------------------------------------------------------------------------
// launcher
// ---------------------------------------------------------------------------
extern "C" void kernel_launch(void* const* d_in, const int* in_sizes, int n_in,
                              void* d_out, int out_size, void* d_ws, size_t ws_size,
                              hipStream_t stream) {
  (void)in_sizes; (void)n_in; (void)out_size; (void)ws_size;

  const float* clip   = (const float*)d_in[0];
  const int*   lens   = (const int*)  d_in[1];
  const float* w_wei  = (const float*)d_in[2];
  const float* b_wei  = (const float*)d_in[3];
  const float* w_mfc  = (const float*)d_in[4];
  const float* b_mfc  = (const float*)d_in[5];
  const float* w_afc  = (const float*)d_in[6];
  const float* b_afc  = (const float*)d_in[7];
  const float* w_pool = (const float*)d_in[8];
  const float* b_pool = (const float*)d_in[9];

  char* ws = (char*)d_ws;
  float*  scores = (float*) (ws + 0);           //  64 KB : B*L f32
  float*  awin   = (float*) (ws + 65536);       // 191 KB : B*LW*3 f32
  __bf16* clipb  = (__bf16*)(ws + 260608);      //  64 MB : B*L*D bf16
  __bf16* vidb   = (__bf16*)(ws + 67369472);    //  64 MB : B*LW*D bf16
  __bf16* wTm    = (__bf16*)(ws + 133954048);   //   4 MB : E*D bf16
  __bf16* wTa    = (__bf16*)(ws + 138148352);   //   4 MB : E*D bf16
  float*  motion = (float*) (ws + 142342656);   //  64 MB : B*LW*E f32
  float*  appear = (float*) (ws + 208927232);   //  64 MB : B*L*E f32

  scores_kernel<<<dim3((BB * LL) / 8), 256, 0, stream>>>(clip, w_wei, b_wei, lens, scores);
  win_softmax_kernel<<<dim3((BB * LW + 255) / 256), 256, 0, stream>>>(scores, awin);
  conv_clip_kernel<<<dim3((BB * LL * (DD / 4)) / 256), 256, 0, stream>>>(clip, clipb);
  conv_vid_kernel<<<dim3((BB * LW * (DD / 4)) / 256), 256, 0, stream>>>(clip, awin, vidb);
  convw_kernel<<<dim3(EE / 32, DD / 32), 256, 0, stream>>>(w_mfc, wTm);
  convw_kernel<<<dim3(EE / 32, DD / 32), 256, 0, stream>>>(w_afc, wTa);
  gemm_bf16_kernel<<<dim3(EE / 128, (BB * LW) / 128), 256, 0, stream>>>(vidb, wTm, b_mfc, motion);
  gemm_bf16_kernel<<<dim3(EE / 128, (BB * LL) / 128), 256, 0, stream>>>(clipb, wTa, b_afc, appear);
  pool_kernel<<<dim3(BB, 2), 256, 0, stream>>>(motion, appear, w_pool, b_pool, lens,
                                               (float*)d_out);
}
